// DeformableAttention_19851338842356
// MI455X (gfx1250) — matrix operand order repaired and checked by reading, not compiled, not run on hardware
//
#include <hip/hip_runtime.h>

// ---------------------------------------------------------------------------
// Deformable attention (1-D) for MI455X / gfx1250.
//  - GEMMs: v_wmma_f32_16x16x32_bf16, f32 accumulate, bf16 inputs
//  - B (weight) panel staged in LDS via the Tensor Data Mover (TDM)
//  - sampling/softmax on VALU; projected value tensor stays L2-resident
// ---------------------------------------------------------------------------

typedef __bf16 bf16;
typedef __attribute__((ext_vector_type(16))) __bf16 v16bf;
typedef __attribute__((ext_vector_type(8)))  __bf16 v8bf;
typedef __attribute__((ext_vector_type(8)))  float   v8f;
typedef __attribute__((ext_vector_type(4)))  float   v4f;
typedef __attribute__((ext_vector_type(4)))  unsigned int u32x4;
typedef __attribute__((ext_vector_type(8)))  int          i32x8;
typedef __attribute__((ext_vector_type(4)))  int          i32x4;

#define E_  1024
#define H_  16
#define D_  64
#define K_  4
#define LQ_ 2048
#define LV_ 4096
#define BZ_ 8

// Padded LDS layout for the B panel: 16B of pad inserted per 1024B of data
// (TDM pad_interval=256 dwords, pad_amount=4 dwords).  For K=1024 bf16 this
// gives a row stride of 2*K + (K/512)*16 = 2080 bytes, i.e. consecutive rows
// are offset by 8 LDS banks -> at most 2-way conflicts on fragment reads.
#define BPANEL_ROWS   64
#define BROW_STRIDE(Kd) (2 * (Kd) + ((Kd) >> 9) * 16)
#define BPANEL_BYTES  (BPANEL_ROWS * 2080)   // for K=1024

// ---------------------------------------------------------------------------
// f32 -> bf16 elementwise conversion (weight matrices)
// ---------------------------------------------------------------------------
__global__ void convert_f32_bf16(const float* __restrict__ in,
                                 bf16* __restrict__ out, int n) {
    int i = blockIdx.x * blockDim.x + threadIdx.x;
    if (i < n) out[i] = (bf16)in[i];
}

// ---------------------------------------------------------------------------
// Generic bf16-WMMA GEMM:  C[m][n] = sum_k A[m][k] * B[n][k] + bias[n]
//   A: f32 row-major [M][K]  (converted to bf16 on the fly)
//   B: bf16 row-major [N][K] (pre-converted weights; 64-row panel per block
//      staged into LDS once via TDM)
//   permuteBZ: if >0, output row m=(lv*BZ+b) is stored at row (b*LVd+lv)
//   mask: optional per-output-row zeroing (key padding mask)
// Block: 256 threads = 8 waves; wave computes a 16(M) x 64(N) strip.
// Macro-tile: 128(M) x 64(N). Requires M%128==0, N%64==0, K%32==0, K<=65535.
// Grid: (N/64, M/128)  -- N fastest so one A band stays hot in L2.
// ---------------------------------------------------------------------------
__launch_bounds__(256)
__global__ void gemm_bf16_wmma(const float* __restrict__ A,
                               const bf16*  __restrict__ B,
                               const float* __restrict__ bias,
                               float* __restrict__ C,
                               int M, int N, int K,
                               int permuteBZ,
                               const unsigned char* __restrict__ mask) {
    extern __shared__ char smem[];       // padded B panel, BPANEL_BYTES
    const int lane = threadIdx.x & 31;   // wave32
    const int wid  = threadIdx.x >> 5;   // 0..7
    const int row  = lane & 15;
    const int hi   = lane >> 4;          // lane half

    const int n0 = blockIdx.x * 64;      // N is the fast grid dimension
    const int m0 = blockIdx.y * 128 + wid * 16;
    const int rstride = BROW_STRIDE(K);

    // ---- stage the 64 x K B panel into LDS --------------------------------
#if __has_builtin(__builtin_amdgcn_tensor_load_to_lds)
    if (wid == 0) {
        // Tensor DMA descriptor (D#), 2-D tensor, 2-byte elements, LDS pad on.
        unsigned long long gaddr =
            (unsigned long long)(uintptr_t)(B + (size_t)n0 * K);
        unsigned int laddr = (unsigned int)(uintptr_t)smem;  // LDS byte offset
        u32x4 g0 = { 1u,                                  // count=1 descriptor
                     laddr,                               // lds_addr
                     (unsigned int)gaddr,                 // global_addr[31:0]
                     ((unsigned int)(gaddr >> 32) & 0x01FFFFFFu)
                         | 0x80000000u };                 // addr[56:32]|type=2
        i32x8 g1 = { (int)0x07D10000,   // data_size=2B, pad_en, intv=256dw, amt=4dw
                     (int)(((unsigned)K & 0xFFFFu) << 16),   // tensor_dim0 lo
                     (int)(((unsigned)N & 0xFFFFu) << 16),   // dim0 hi | dim1 lo
                     (int)(((unsigned)K & 0xFFFFu) << 16),   // dim1 hi | tile_dim0=K
                     BPANEL_ROWS,                            // tile_dim1=64
                     K,                                      // dim0_stride lo
                     0, 0 };
        i32x4 gz = { 0, 0, 0, 0 };
#if defined(__clang_major__) && __clang_major__ >= 23
        i32x8 gz8 = { 0, 0, 0, 0, 0, 0, 0, 0 };
        __builtin_amdgcn_tensor_load_to_lds(g0, g1, gz, gz, gz8, 0);
#else
        __builtin_amdgcn_tensor_load_to_lds(g0, g1, gz, gz, 0);
#endif
        __builtin_amdgcn_s_wait_tensorcnt(0);
    }
#else
    // Fallback: cooperative copy with the identical padded layout.
    for (int idx = threadIdx.x * 8; idx < BPANEL_ROWS * K; idx += 256 * 8) {
        const int r = idx / K, e = idx % K;
        const bf16* src = B + (size_t)(n0 + r) * K + e;
        char* dst = smem + r * rstride + 2 * e + ((e >> 9) << 4);
        *(v8bf*)dst = *(const v8bf*)src;
    }
#endif
    __syncthreads();

    v8f acc[4] = { (v8f){}, (v8f){}, (v8f){}, (v8f){} };

    // A fragment: lane (row,hi) holds row m0+row, contiguous K-elements
    // {hi*16..+15} (matches the B fragment k order).
    const float* arow = A + (size_t)(m0 + row) * K + hi * 16;
    // B fragment row bases in LDS (column n0+j*16+row of B).
    const char* bb[4];
#pragma unroll
    for (int j = 0; j < 4; ++j) bb[j] = smem + (j * 16 + row) * rstride;

    for (int k0 = 0; k0 < K; k0 += 32) {
        // ---- load + convert A fragment (4 x b128 loads, packed cvt) ----
        v4f a0 = *(const v4f*)(arow + k0);
        v4f a1 = *(const v4f*)(arow + k0 + 4);
        v4f a2 = *(const v4f*)(arow + k0 + 8);
        v4f a3 = *(const v4f*)(arow + k0 + 12);
        v16bf afrag;
#pragma unroll
        for (int i = 0; i < 4; ++i) {
            afrag[i]      = (bf16)a0[i];
            afrag[i + 4]  = (bf16)a1[i];
            afrag[i + 8]  = (bf16)a2[i];
            afrag[i + 12] = (bf16)a3[i];
        }
        // ---- B fragments from LDS (2 x ds_load_b128 per tile) ----
        // lane chunk: 16 contiguous K-elements starting at e = k0 + hi*16
        const int e    = k0 + hi * 16;
        const int boff = 2 * e + ((e >> 9) << 4);   // pad-adjusted byte offset
        v16bf bfr[4];
#pragma unroll
        for (int j = 0; j < 4; ++j) {
            v8bf lo = *(const v8bf*)(bb[j] + boff);
            v8bf hh = *(const v8bf*)(bb[j] + boff + 16);
            bfr[j] = __builtin_shufflevector(lo, hh, 0, 1, 2, 3, 4, 5, 6, 7,
                                             8, 9, 10, 11, 12, 13, 14, 15);
        }
#pragma unroll
        for (int j = 0; j < 4; ++j)
            acc[j] = __builtin_amdgcn_wmma_f32_16x16x32_bf16(
                false, afrag, false, bfr[j], (short)0, acc[j], false, false);
    }

    // ---- epilogue: bias + optional mask + optional (lv,b)->(b,lv) permute --
    const int LVd = (permuteBZ > 0) ? (M / permuteBZ) : 0;
#pragma unroll
    for (int j = 0; j < 4; ++j) {
        const int n  = n0 + j * 16 + row;   // C/D: N = lane&15
        const float bs = bias ? bias[n] : 0.0f;
#pragma unroll
        for (int r = 0; r < 8; ++r) {
            const int m = m0 + 8 * hi + r;  // C/D: M = r (+8 for lane half 1)
            int outRow = m;
            if (permuteBZ == 8) {           // fast path (BZ=8)
                outRow = (m & 7) * LVd + (m >> 3);
            } else if (permuteBZ > 0) {
                outRow = (m % permuteBZ) * LVd + (m / permuteBZ);
            }
            float val = acc[j][r] + bs;
            if (mask && mask[outRow]) val = 0.0f;
            C[(size_t)outRow * N + n] = val;
        }
    }
}

// ---------------------------------------------------------------------------
// Deformable sampler: softmax over K=4, bilinear 1-D gather from the
// projected value tensor (layout [BZ][LV][E], L2-resident), weighted sum.
// One 64-thread unit per (q,b,h); 4 units per 256-thread block.
// ---------------------------------------------------------------------------
__launch_bounds__(256)
__global__ void deform_sampler(const float* __restrict__ vproj,   // [BZ][LV][E]
                               const float* __restrict__ offraw,  // [LQ*BZ][H*K]
                               const float* __restrict__ awraw,   // [LQ*BZ][H*K]
                               const float* __restrict__ refp,    // [BZ][LQ]
                               const float* __restrict__ snip,    // [BZ]
                               float* __restrict__ attn) {        // [LQ*BZ][E]
    const int d = threadIdx.x & 63;
    const int u = blockIdx.x * 4 + (threadIdx.x >> 6);  // (q*BZ+b)*H + h
    const int h = u % H_;
    const int t = u / H_;        // q*BZ + b
    const int b = t % BZ_;
    const int q = t / BZ_;

    const float* orow = offraw + (size_t)t * (H_ * K_) + h * K_;
    const float* arow = awraw  + (size_t)t * (H_ * K_) + h * K_;
    const float ref   = refp[b * LQ_ + q];
    const float snInv = 1.0f / snip[b];

    float aw[K_];
#pragma unroll
    for (int k = 0; k < K_; ++k) aw[k] = arow[k];
    float mx = fmaxf(fmaxf(aw[0], aw[1]), fmaxf(aw[2], aw[3]));
    float e[K_]; float esum = 0.0f;
#pragma unroll
    for (int k = 0; k < K_; ++k) { e[k] = __expf(aw[k] - mx); esum += e[k]; }
    const float einv = 1.0f / esum;

    const float* vbase = vproj + ((size_t)b * LV_) * E_ + h * D_ + d;

    float acc = 0.0f;
#pragma unroll
    for (int k = 0; k < K_; ++k) {
        const float pos = ref + orow[k] * snInv;        // in [0,1]
        const float x   = pos * (float)(LV_ - 1);       // align_corners=True
        const float x0f = floorf(x);
        const float fr  = x - x0f;
        const int i0 = (int)x0f;
        const int i1 = i0 + 1;
        const float w0 = (i0 >= 0 && i0 < LV_) ? (1.0f - fr) : 0.0f;
        const float w1 = (i1 >= 0 && i1 < LV_) ? fr : 0.0f;
        const int i0c = min(max(i0, 0), LV_ - 1);
        const int i1c = min(max(i1, 0), LV_ - 1);
        const float g0 = vbase[(size_t)i0c * E_];
        const float g1 = vbase[(size_t)i1c * E_];
        acc += (e[k] * einv) * (w0 * g0 + w1 * g1);
    }
    attn[(size_t)t * E_ + h * D_ + d] = acc;
}

// ---------------------------------------------------------------------------
extern "C" void kernel_launch(void* const* d_in, const int* in_sizes, int n_in,
                              void* d_out, int out_size, void* d_ws, size_t ws_size,
                              hipStream_t stream) {
    const float* query = (const float*)d_in[0];          // [LQ][BZ][E]
    const float* value = (const float*)d_in[1];          // [LV][BZ][E]
    const unsigned char* mask = (const unsigned char*)d_in[2]; // [BZ][LV] bool
    const float* refp  = (const float*)d_in[3];          // [BZ][LQ][1]
    const float* snip  = (const float*)d_in[4];          // [BZ]
    const float* Wv    = (const float*)d_in[5];
    const float* bv    = (const float*)d_in[6];
    const float* Wo    = (const float*)d_in[7];
    const float* bo    = (const float*)d_in[8];
    const float* Wa    = (const float*)d_in[9];
    const float* ba    = (const float*)d_in[10];
    const float* Wout  = (const float*)d_in[11];
    const float* bout  = (const float*)d_in[12];
    float* out = (float*)d_out;

    // ---- workspace carving (all sizes multiples of 256B) ----
    char* ws = (char*)d_ws;
    size_t off = 0;
    auto carve = [&](size_t bytes) -> void* {
        void* p = ws + off;
        off += (bytes + 255) & ~(size_t)255;
        return p;
    };
    bf16*  Wv_bf   = (bf16*)carve((size_t)E_ * E_ * sizeof(bf16));
    bf16*  Wo_bf   = (bf16*)carve((size_t)(H_ * K_) * E_ * sizeof(bf16));
    bf16*  Wa_bf   = (bf16*)carve((size_t)(H_ * K_) * E_ * sizeof(bf16));
    bf16*  Wout_bf = (bf16*)carve((size_t)E_ * E_ * sizeof(bf16));
    float* vproj   = (float*)carve((size_t)BZ_ * LV_ * E_ * sizeof(float));
    float* offraw  = (float*)carve((size_t)LQ_ * BZ_ * (H_ * K_) * sizeof(float));
    float* awraw   = (float*)carve((size_t)LQ_ * BZ_ * (H_ * K_) * sizeof(float));
    float* attn    = (float*)carve((size_t)LQ_ * BZ_ * E_ * sizeof(float));

    const size_t smem_bytes = BPANEL_BYTES;   // 133,120 B (<= 320KB/WGP)

    // ---- 1. pre-convert weights to bf16 ----
    {
        int n1 = E_ * E_, n2 = H_ * K_ * E_;
        convert_f32_bf16<<<(n1 + 255) / 256, 256, 0, stream>>>(Wv,   Wv_bf,   n1);
        convert_f32_bf16<<<(n2 + 255) / 256, 256, 0, stream>>>(Wo,   Wo_bf,   n2);
        convert_f32_bf16<<<(n2 + 255) / 256, 256, 0, stream>>>(Wa,   Wa_bf,   n2);
        convert_f32_bf16<<<(n1 + 255) / 256, 256, 0, stream>>>(Wout, Wout_bf, n1);
    }

    // ---- 2. value projection: [LV*BZ, E] x [E, E]^T -> vproj[BZ][LV][E] ----
    {
        const int M = LV_ * BZ_, N = E_, Kd = E_;
        dim3 grid(N / 64, M / 128);           // N fastest: A band hot in L2
        gemm_bf16_wmma<<<grid, 256, smem_bytes, stream>>>(
            value, Wv_bf, bv, vproj, M, N, Kd, BZ_, mask);
    }

    // ---- 3. sampling offsets & attention logits: [LQ*BZ, E] x [64, E]^T ----
    {
        const int M = LQ_ * BZ_, N = H_ * K_, Kd = E_;
        dim3 grid(N / 64, M / 128);
        gemm_bf16_wmma<<<grid, 256, smem_bytes, stream>>>(
            query, Wo_bf, bo, offraw, M, N, Kd, 0, nullptr);
        gemm_bf16_wmma<<<grid, 256, smem_bytes, stream>>>(
            query, Wa_bf, ba, awraw, M, N, Kd, 0, nullptr);
    }

    // ---- 4. softmax + bilinear grid-sample + weighted sum ----
    {
        const int units = LQ_ * BZ_ * H_;       // 262144
        deform_sampler<<<units / 4, 256, 0, stream>>>(vproj, offraw, awraw,
                                                      refp, snip, attn);
    }

    // ---- 5. output projection: [LQ*BZ, E] x [E, E]^T -> out ----
    {
        const int M = LQ_ * BZ_, N = E_, Kd = E_;
        dim3 grid(N / 64, M / 128);
        gemm_bf16_wmma<<<grid, 256, smem_bytes, stream>>>(
            attn, Wout_bf, bout, out, M, N, Kd, 0, nullptr);
    }
}
